// SimpleAttention_197568496157
// MI455X (gfx1250) — compile-verified, matrix operand order checked by reading, not run on hardware
//
#include <hip/hip_runtime.h>
#include <hip/hip_bf16.h>

// ---------------------------------------------------------------------------
// CDNA5 (gfx1250) fused causal attention: QKV GEMM + RoPE -> flash attention
// -> output projection.  All matmuls via v_wmma_f32_16x16x32_bf16 (wave32).
// bf16 intermediates in workspace; double-buffered LDS pipelines; K/V and
// O-proj A tiles staged with GLOBAL_LOAD_ASYNC_TO_LDS_B128 (ASYNCcnt).
// f32->bf16 via __bf16 fptrunc (RNE) so the backend can select hardware cvt.
// ---------------------------------------------------------------------------

typedef __attribute__((ext_vector_type(16))) __bf16        v16bf;
typedef __attribute__((ext_vector_type(2)))  __bf16        v2bf;
typedef __attribute__((ext_vector_type(8)))  float         v8f;
typedef __attribute__((ext_vector_type(4)))  unsigned int  u32x4;
typedef __attribute__((ext_vector_type(4)))  float         f32x4;

union Frag {
  v16bf v;
  u32x4 q[2];
};

// fptrunc f32->bf16 is round-to-nearest-even (matches JAX astype).
__device__ __forceinline__ unsigned short f2bf(float f) {
  union { __bf16 h; unsigned short u; } c;
  c.h = (__bf16)f;
  return c.u;
}
// build_vector of two fptruncs -> lets the backend select v_cvt_pk_bf16_f32.
__device__ __forceinline__ unsigned int pack2(float lo, float hi) {
  union { v2bf v; unsigned int u; } c;
  c.v[0] = (__bf16)lo;
  c.v[1] = (__bf16)hi;
  return c.u;
}

// Async DMA: LDS[lds_off] = MEM[g .. g+15] per lane, tracked by ASYNCcnt.
__device__ __forceinline__ void async_ld_b128(unsigned int lds_off, const void* g) {
  asm volatile("global_load_async_to_lds_b128 %0, %1, off"
               :: "v"(lds_off), "v"(g) : "memory");
}
__device__ __forceinline__ void wait_async0() {
  asm volatile("s_wait_asynccnt 0x0" ::: "memory");
}
__device__ __forceinline__ unsigned int lds_off(const void* p) {
  return (unsigned int)(size_t)p;    // flat LDS address: low 32 bits = LDS offset
}

// Load a 16(rows) x 32(K) bf16 fragment from a row-major tile, row stride `ld`
// (halves).  ISA 16-bit A 16x32 layout: lane m = L&15, h = L>>4; VGPR0..3 <-
// K = h*8..+7, VGPR4..7 <- K = 16+h*8..+7.  Same loader yields B fragments
// from row-major [N,K] tiles by the A/B transpose duality.
__device__ __forceinline__ void load_frag(Frag& f, const unsigned short* base,
                                          int ld, int lane) {
  const int m = lane & 15, h = lane >> 4;
  const unsigned short* r0 = base + m * ld + h * 8;
  f.q[0] = *(const u32x4*)(r0);
  f.q[1] = *(const u32x4*)(r0 + 16);
}

#define WMMA_BF16(A, B, C) \
  __builtin_amdgcn_wmma_f32_16x16x32_bf16(false, (A), false, (B), (short)0, (C), false, false)

// ---------------------------------------------------------------------------
// Kernel 1: qkv = x @ Wqkv^T  (M=4096, N=3072, K=1024), fused RoPE epilogue.
// 128x64 workgroup tile, 8 waves x (2x2 WMMA tiles).  Double-buffered LDS,
// one barrier per K-step; next tile's f32 loads overlap current WMMAs.
// Writes Q,K bf16 [bh][s][64] and V transposed [bh][64][s].
// ---------------------------------------------------------------------------
__global__ __launch_bounds__(256) void qkv_rope_kernel(
    const float* __restrict__ x, const float* __restrict__ Wqkv,
    unsigned short* __restrict__ Qw, unsigned short* __restrict__ Kw,
    unsigned short* __restrict__ Vt)
{
  constexpr int S = 2048, D = 1024, H = 16;
  __shared__ alignas(16) unsigned int lds_a[2][128 * 20];  // 128x32 bf16, pad->40 halves
  __shared__ alignas(16) unsigned int lds_b[2][64 * 20];
  __shared__ float stage[128 * 65];

  const int tid  = threadIdx.x;
  const int lane = tid & 31, wid = tid >> 5;
  const int waveM = wid >> 1, waveN = wid & 1;
  const int m0 = blockIdx.x * 128;
  const int n0 = blockIdx.y * 64;

  v8f acc[2][2] = {};
  const int ar = tid >> 1, ah = tid & 1;   // A loader: row, 16-col half
  const int br = tid >> 2, bq = tid & 3;   // B loader: row, 8-col quarter

  f32x4 ra[4], rb[2];
  auto ld_regs = [&](int k0) {
    const f32x4* p = (const f32x4*)(x + (size_t)(m0 + ar) * D + k0 + ah * 16);
    ra[0] = p[0]; ra[1] = p[1]; ra[2] = p[2]; ra[3] = p[3];
    const f32x4* q = (const f32x4*)(Wqkv + (size_t)(n0 + br) * D + k0 + bq * 8);
    rb[0] = q[0]; rb[1] = q[1];
  };
  auto st_tile = [&](int buf) {
    unsigned int* d = &lds_a[buf][ar * 20 + ah * 8];
    d[0]=pack2(ra[0].x,ra[0].y); d[1]=pack2(ra[0].z,ra[0].w);
    d[2]=pack2(ra[1].x,ra[1].y); d[3]=pack2(ra[1].z,ra[1].w);
    d[4]=pack2(ra[2].x,ra[2].y); d[5]=pack2(ra[2].z,ra[2].w);
    d[6]=pack2(ra[3].x,ra[3].y); d[7]=pack2(ra[3].z,ra[3].w);
    unsigned int* e = &lds_b[buf][br * 20 + bq * 4];
    e[0]=pack2(rb[0].x,rb[0].y); e[1]=pack2(rb[0].z,rb[0].w);
    e[2]=pack2(rb[1].x,rb[1].y); e[3]=pack2(rb[1].z,rb[1].w);
  };

  ld_regs(0); st_tile(0);
  for (int it = 0; it < 32; ++it) {
    const int cur = it & 1;
    __syncthreads();                       // buffer `cur` ready (all waves)
    const bool has = (it + 1) < 32;
    if (has) ld_regs(it * 32 + 32);        // global f32 loads in flight over WMMAs
    Frag a0, a1, b0, b1;
    const unsigned short* la = (const unsigned short*)lds_a[cur];
    const unsigned short* lb = (const unsigned short*)lds_b[cur];
    load_frag(a0, la + (waveM * 32     ) * 40, 40, lane);
    load_frag(a1, la + (waveM * 32 + 16) * 40, 40, lane);
    load_frag(b0, lb + (waveN * 32     ) * 40, 40, lane);
    load_frag(b1, lb + (waveN * 32 + 16) * 40, 40, lane);
    acc[0][0] = WMMA_BF16(a0.v, b0.v, acc[0][0]);
    acc[0][1] = WMMA_BF16(a0.v, b1.v, acc[0][1]);
    acc[1][0] = WMMA_BF16(a1.v, b0.v, acc[1][0]);
    acc[1][1] = WMMA_BF16(a1.v, b1.v, acc[1][1]);
    if (has) st_tile(cur ^ 1);
  }

  // Stage C to LDS: lane holds col n for 8 rows per accumulator VGPR.
  const int cn = lane & 15, ch = lane >> 4;
  #pragma unroll
  for (int mt = 0; mt < 2; ++mt)
    #pragma unroll
    for (int nt = 0; nt < 2; ++nt)
      #pragma unroll
      for (int r = 0; r < 8; ++r)
        stage[(waveM*32 + mt*16 + ch*8 + r) * 65 + waveN*32 + nt*16 + cn] = acc[mt][nt][r];
  __syncthreads();

  // Epilogue: partial RoPE (16 live freqs of half=32) for Q,K; V stored transposed.
  for (int it = 0; it < 32; ++it) {
    const int e = it * 256 + tid;
    const int r = e >> 6, c = e & 63;
    const int gm = m0 + r;
    const int b  = gm >> 11, s = gm & 2047;
    const int gn = n0 + c;
    const int which = gn >> 10;          // 0=q, 1=k, 2=v
    const int rem = gn & 1023;
    const int h = rem >> 6, dki = rem & 63;
    if (which < 2) {
      const int j = (dki < 32) ? dki : dki - 32;
      const float x1 = stage[r * 65 + j];
      const float x2 = stage[r * 65 + j + 32];
      const float ang = (j < 16) ? __expf(-0.6140226915f * (float)j) : 0.0f;
      const float th = (float)s * ang;
      const float cth = __cosf(th), sth = __sinf(th);
      const float o = (dki < 32) ? (x1 * cth - x2 * sth) : (x1 * sth + x2 * cth);
      unsigned short* dst = (which == 0) ? Qw : Kw;
      dst[((size_t)(b * H + h) * S + s) * 64 + dki] = f2bf(o);
    } else {
      Vt[((size_t)(b * H + h) * 64 + dki) * S + s] = f2bf(stage[r * 65 + c]);
    }
  }
}

// ---------------------------------------------------------------------------
// Kernel 2: causal flash attention.  grid = (S/128, B*H); 8 waves, each wave
// owns 16 query rows.  K/V tiles (4 KB each per 32-key step) cooperatively
// staged into double-buffered LDS with async-to-LDS DMA: copy of tile t+1
// overlaps softmax + P.V WMMAs of tile t.
// ---------------------------------------------------------------------------
__global__ __launch_bounds__(256) void flash_attn_kernel(
    const unsigned short* __restrict__ Qw, const unsigned short* __restrict__ Kw,
    const unsigned short* __restrict__ Vt, unsigned short* __restrict__ Ow)
{
  constexpr int S = 2048, DK = 64;
  __shared__ alignas(16) unsigned short ktile[2][32 * 72];  // 32 keys x 64 dk, pad->72
  __shared__ alignas(16) unsigned short vtile[2][64 * 40];  // 64 dims x 32 keys, pad->40
  __shared__ alignas(16) unsigned short ptile[8][16 * 40];  // per-wave P restage

  const int tid = threadIdx.x, lane = tid & 31, wid = tid >> 5;
  const int n = lane & 15, hh = lane >> 4;
  const int bh = blockIdx.y;
  const int b = bh >> 4, h = bh & 15;
  const int q0 = blockIdx.x * 128;
  const int qw = q0 + wid * 16;

  const unsigned short* Qp = Qw + ((size_t)bh * S + qw) * DK;
  const unsigned short* Kp = Kw + (size_t)bh * S * DK;
  const unsigned short* Vp = Vt + (size_t)bh * DK * S;

  Frag aq0, aq1;
  load_frag(aq0, Qp,      DK, lane);
  load_frag(aq1, Qp + 32, DK, lane);

  // cooperative async staging of one 32-key K/V tile (256 threads, 16B chunks)
  const int kr = tid >> 3, kc = tid & 7;   // K: 32 rows x 8 chunks
  const int vr = tid >> 2, vc = tid & 3;   // V: 64 rows x 4 chunks
  auto stage_kv = [&](int buf, int kb) {
    async_ld_b128(lds_off(&ktile[buf][kr * 72 + kc * 8]),
                  Kp + (size_t)(kb + kr) * DK + kc * 8);
    async_ld_b128(lds_off(&vtile[buf][vr * 40 + vc * 8]),
                  Vp + (size_t)vr * S + kb + vc * 8);
  };

  float m_r[8], l_r[8];
  #pragma unroll
  for (int r = 0; r < 8; ++r) { m_r[r] = -1e30f; l_r[r] = 0.0f; }
  v8f accv[4] = {};

  const int nst = (q0 + 128) >> 5;     // uniform across workgroup (barrier-safe)
  stage_kv(0, 0);

  for (int st = 0; st < nst; ++st) {
    const int kb = st * 32;
    const int cur = st & 1;
    wait_async0();                     // own chunks of `cur` done
    __syncthreads();                   // everyone's chunks done; prev reads retired
    if (st + 1 < nst) stage_kv(cur ^ 1, kb + 32);   // DMA overlaps compute below

    const unsigned short* kp = ktile[cur];
    Frag bk;
    v8f s1 = {}, s2 = {};
    load_frag(bk, kp,               72, lane);  s1 = WMMA_BF16(aq0.v, bk.v, s1);
    load_frag(bk, kp + 32,          72, lane);  s1 = WMMA_BF16(aq1.v, bk.v, s1);
    load_frag(bk, kp + 16 * 72,     72, lane);  s2 = WMMA_BF16(aq0.v, bk.v, s2);
    load_frag(bk, kp + 16 * 72 + 32, 72, lane); s2 = WMMA_BF16(aq1.v, bk.v, s2);

    const int k1 = kb + n, k2 = kb + 16 + n;
    float e1[8], e2[8], f_[8];
    #pragma unroll
    for (int r = 0; r < 8; ++r) {
      const int qi = qw + hh * 8 + r;
      float t1 = (k1 <= qi) ? s1[r] * 0.125f : -1e30f;   // scale = 1/sqrt(64)
      float t2 = (k2 <= qi) ? s2[r] * 0.125f : -1e30f;
      float lm = fmaxf(t1, t2);
      lm = fmaxf(lm, __shfl_xor(lm, 1, 32));
      lm = fmaxf(lm, __shfl_xor(lm, 2, 32));
      lm = fmaxf(lm, __shfl_xor(lm, 4, 32));
      lm = fmaxf(lm, __shfl_xor(lm, 8, 32));   // row max over 16 key-lanes
      const float mn = fmaxf(m_r[r], lm);
      const float fr = __expf(m_r[r] - mn);
      const float p1 = __expf(t1 - mn);
      const float p2 = __expf(t2 - mn);
      float rs = p1 + p2;
      rs += __shfl_xor(rs, 1, 32);
      rs += __shfl_xor(rs, 2, 32);
      rs += __shfl_xor(rs, 4, 32);
      rs += __shfl_xor(rs, 8, 32);
      l_r[r] = l_r[r] * fr + rs;
      m_r[r] = mn;
      f_[r] = fr; e1[r] = p1; e2[r] = p2;
    }
    #pragma unroll
    for (int nt = 0; nt < 4; ++nt)
      #pragma unroll
      for (int r = 0; r < 8; ++r)
        accv[nt][r] *= f_[r];

    // Restage P (16x32) from C-layout into A-fragment layout via LDS.
    // One packed cvt per (e1,e2) pair; low half -> col n, high -> col 16+n.
    unsigned short* pw = ptile[wid];
    #pragma unroll
    for (int r = 0; r < 8; ++r) {
      const unsigned int pu = pack2(e1[r], e2[r]);
      pw[(hh * 8 + r) * 40 + n]      = (unsigned short)pu;
      pw[(hh * 8 + r) * 40 + 16 + n] = (unsigned short)(pu >> 16);
    }
    __syncthreads();                   // fence: P stores -> fragment loads
    Frag pA;
    load_frag(pA, pw, 40, lane);
    #pragma unroll
    for (int nt = 0; nt < 4; ++nt) {
      Frag bv;
      load_frag(bv, vtile[cur] + nt * 16 * 40, 40, lane);
      accv[nt] = WMMA_BF16(pA.v, bv.v, accv[nt]);
    }
  }

  // out[b][s][h*64+dk] = acc / l   (bf16, feeds the O-projection GEMM)
  #pragma unroll
  for (int nt = 0; nt < 4; ++nt)
    #pragma unroll
    for (int r = 0; r < 8; ++r) {
      const int s_idx = qw + hh * 8 + r;
      const float o = accv[nt][r] / l_r[r];
      Ow[((size_t)b * S + s_idx) * 1024 + h * 64 + nt * 16 + n] = f2bf(o);
    }
}

// ---------------------------------------------------------------------------
// Kernel 3: out = O @ Wo^T  (M=4096, N=1024, K=1024), fp32 output.
// A (bf16) staged with async-to-LDS DMA; B register-staged (needs f32->bf16).
// Double-buffered, one barrier per K-step.
// ---------------------------------------------------------------------------
__global__ __launch_bounds__(256) void oproj_kernel(
    const unsigned short* __restrict__ Ow, const float* __restrict__ Wo,
    float* __restrict__ out)
{
  constexpr int D = 1024;
  __shared__ alignas(16) unsigned int lds_a[2][128 * 20];
  __shared__ alignas(16) unsigned int lds_b[2][64 * 20];

  const int tid = threadIdx.x, lane = tid & 31, wid = tid >> 5;
  const int waveM = wid >> 1, waveN = wid & 1;
  const int m0 = blockIdx.x * 128, n0 = blockIdx.y * 64;
  v8f acc[2][2] = {};
  const int br = tid >> 2, bq = tid & 3;

  auto stage_a = [&](int buf, int k0) {       // 128x32 halves = 512 x 16B chunks
    #pragma unroll
    for (int i = 0; i < 2; ++i) {
      const int ch = tid + i * 256;
      const int r = ch >> 2, c = ch & 3;
      async_ld_b128(lds_off(&lds_a[buf][r * 20 + c * 4]),
                    Ow + (size_t)(m0 + r) * D + k0 + c * 8);
    }
  };
  f32x4 rb[2];
  auto ld_b = [&](int k0) {
    const f32x4* q = (const f32x4*)(Wo + (size_t)(n0 + br) * D + k0 + bq * 8);
    rb[0] = q[0]; rb[1] = q[1];
  };
  auto st_b = [&](int buf) {
    unsigned int* e = &lds_b[buf][br * 20 + bq * 4];
    e[0]=pack2(rb[0].x,rb[0].y); e[1]=pack2(rb[0].z,rb[0].w);
    e[2]=pack2(rb[1].x,rb[1].y); e[3]=pack2(rb[1].z,rb[1].w);
  };

  stage_a(0, 0); ld_b(0); st_b(0);
  for (int it = 0; it < 32; ++it) {
    const int cur = it & 1;
    wait_async0();
    __syncthreads();
    const bool has = (it + 1) < 32;
    if (has) { stage_a(cur ^ 1, it * 32 + 32); ld_b(it * 32 + 32); }
    Frag a0, a1, b0, b1;
    const unsigned short* la = (const unsigned short*)lds_a[cur];
    const unsigned short* lb = (const unsigned short*)lds_b[cur];
    load_frag(a0, la + (waveM * 32     ) * 40, 40, lane);
    load_frag(a1, la + (waveM * 32 + 16) * 40, 40, lane);
    load_frag(b0, lb + (waveN * 32     ) * 40, 40, lane);
    load_frag(b1, lb + (waveN * 32 + 16) * 40, 40, lane);
    acc[0][0] = WMMA_BF16(a0.v, b0.v, acc[0][0]);
    acc[0][1] = WMMA_BF16(a0.v, b1.v, acc[0][1]);
    acc[1][0] = WMMA_BF16(a1.v, b0.v, acc[1][0]);
    acc[1][1] = WMMA_BF16(a1.v, b1.v, acc[1][1]);
    if (has) st_b(cur ^ 1);
  }

  const int cn = lane & 15, ch = lane >> 4;
  #pragma unroll
  for (int mt = 0; mt < 2; ++mt)
    #pragma unroll
    for (int nt = 0; nt < 2; ++nt)
      #pragma unroll
      for (int r = 0; r < 8; ++r)
        out[(size_t)(m0 + waveM*32 + mt*16 + ch*8 + r) * D
            + n0 + waveN*32 + nt*16 + cn] = acc[mt][nt][r];
}

// ---------------------------------------------------------------------------
extern "C" void kernel_launch(void* const* d_in, const int* in_sizes, int n_in,
                              void* d_out, int out_size, void* d_ws, size_t ws_size,
                              hipStream_t stream) {
  const float* x    = (const float*)d_in[0];
  const float* Wqkv = (const float*)d_in[1];
  const float* Wo   = (const float*)d_in[2];
  // d_in[3] = n_heads (hardcoded 16 to match reference shapes)
  (void)in_sizes; (void)n_in; (void)out_size; (void)ws_size;

  // Workspace layout (bf16 halves): Q | K | V^T | attn-out, 8 MiB each = 32 MiB
  unsigned short* ws = (unsigned short*)d_ws;
  unsigned short* Qw = ws;
  unsigned short* Kw = ws + (size_t)4194304;
  unsigned short* Vt = ws + (size_t)8388608;
  unsigned short* Ow = ws + (size_t)12582912;
  float* out = (float*)d_out;

  qkv_rope_kernel  <<<dim3(32, 48), 256, 0, stream>>>(x, Wqkv, Qw, Kw, Vt);
  flash_attn_kernel<<<dim3(16, 32), 256, 0, stream>>>(Qw, Kw, Vt, Ow);
  oproj_kernel     <<<dim3(32, 16), 256, 0, stream>>>(Ow, Wo, out);
}
